// GATSampling_27376121545166
// MI455X (gfx1250) — matrix operand order repaired and checked by reading, not compiled, not run on hardware
//
#include <hip/hip_runtime.h>
#include <hip/hip_bf16.h>
#include <math.h>

typedef __attribute__((ext_vector_type(16))) _Float16 v16h;
typedef __attribute__((ext_vector_type(8)))  _Float16 v8h;
typedef __attribute__((ext_vector_type(8)))  float    v8f;

#define F_IN   128
#define HD     128
#define NHEAD  4
#define DHEAD  32
#define LDSP   136   // f16 pitch for W^T rows in LDS (272B row stride -> 4-bank lane stride)

// ---------------------------------------------------------------------------
// Out[N,128] = X[N,128] @ W[128,128]; f32 in/out, f16 WMMA with f32 accum.
// One wave computes a 16-row x 128-col stripe; block = 8 waves = 128 rows.
// ---------------------------------------------------------------------------
__global__ __launch_bounds__(256) void gemm128_wmma(
    const float* __restrict__ X, const float* __restrict__ W,
    float* __restrict__ Out, int N)
{
  __shared__ _Float16 wt[128 * LDSP];

  const int tid = threadIdx.x;
  // Stage W^T into LDS as f16: wt[n*LDSP + k] = (f16)W[k*128 + n]
  #pragma unroll 4
  for (int i = tid; i < 128 * 128; i += 256) {
    const int k = i >> 7, n = i & 127;
    wt[n * LDSP + k] = (_Float16)W[i];
  }
  __syncthreads();

  const int wave    = tid >> 5;
  const int lane    = tid & 31;
  const int rowTile = blockIdx.x * 128 + wave * 16;
  if (rowTile >= N) return;                      // N is a multiple of 16

  const int laneLo = lane & 15;
  const int laneHi = (lane >> 4) & 1;
  const int mrow   = rowTile + laneLo;

  v8f acc[8];
  #pragma unroll
  for (int t = 0; t < 8; ++t) acc[t] = (v8f)0.0f;

  const float* __restrict__ xrow = X + (size_t)mrow * F_IN;

  #pragma unroll
  for (int kk = 0; kk < 4; ++kk) {
    // A fragment (16x32 f16): lane row = mrow; element e -> K = kk*32 +
    // (e>>3)*16 + laneHi*8 + (e&7)  => two contiguous 8-float runs.
    const float* p = xrow + kk * 32 + laneHi * 8;
    const float4 q0 = *(const float4*)(p + 0);
    const float4 q1 = *(const float4*)(p + 4);
    const float4 q2 = *(const float4*)(p + 16);
    const float4 q3 = *(const float4*)(p + 20);
    v16h a;
    a[0]  = (_Float16)q0.x; a[1]  = (_Float16)q0.y; a[2]  = (_Float16)q0.z; a[3]  = (_Float16)q0.w;
    a[4]  = (_Float16)q1.x; a[5]  = (_Float16)q1.y; a[6]  = (_Float16)q1.z; a[7]  = (_Float16)q1.w;
    a[8]  = (_Float16)q2.x; a[9]  = (_Float16)q2.y; a[10] = (_Float16)q2.z; a[11] = (_Float16)q2.w;
    a[12] = (_Float16)q3.x; a[13] = (_Float16)q3.y; a[14] = (_Float16)q3.z; a[15] = (_Float16)q3.w;

    #pragma unroll
    for (int nt = 0; nt < 8; ++nt) {
      // B fragment (32x16 f16): lane col = nt*16+laneLo; K = kk*32 + (e>>3)*16 + laneHi*8 + (e&7)
      const _Float16* bp = &wt[(nt * 16 + laneLo) * LDSP + kk * 32 + laneHi * 8];
      const v8h b0 = *(const v8h*)(bp);
      const v8h b1 = *(const v8h*)(bp + 16);
      const v16h b = __builtin_shufflevector(b0, b1,
                       0,1,2,3,4,5,6,7,8,9,10,11,12,13,14,15);
      acc[nt] = __builtin_amdgcn_wmma_f32_16x16x32_f16(
          false, a, false, b, (short)0, acc[nt], false, false);
    }
  }

  // C/D layout: VGPR v, lane L -> row = v + 8*laneHi, col = laneLo
  const int rbase = rowTile + laneHi * 8;
  #pragma unroll
  for (int nt = 0; nt < 8; ++nt) {
    float* op = Out + (size_t)rbase * HD + nt * 16 + laneLo;
    #pragma unroll
    for (int v = 0; v < 8; ++v)
      op[(size_t)v * HD] = acc[nt][v];
  }
}

// ---------------------------------------------------------------------------
// out[row,h] = sum_d F[g(row), h*32+d] * av[h*32+d]; one wave per (row,head).
// gather == nullptr -> identity.
// ---------------------------------------------------------------------------
__global__ __launch_bounds__(128) void dot_heads(
    const float* __restrict__ F, const float* __restrict__ av,
    float* __restrict__ out, const int* __restrict__ gather, int n)
{
  const int row = blockIdx.x;
  if (row >= n) return;
  const int h = threadIdx.x >> 5, lane = threadIdx.x & 31;
  const int srow = gather ? gather[row] : row;
  float v = F[(size_t)srow * HD + h * DHEAD + lane] * av[h * DHEAD + lane];
  #pragma unroll
  for (int off = 16; off; off >>= 1) v += __shfl_xor(v, off, 32);
  if (lane == 0) out[row * NHEAD + h] = v;
}

// ---------------------------------------------------------------------------
// denom[dst,h] += exp(leaky_relu(el[src,h] + er[dst,h]))
// ---------------------------------------------------------------------------
__global__ void edge_softmax_denom(
    const int* __restrict__ src, const int* __restrict__ dst,
    const float* __restrict__ el, const float* __restrict__ er,
    float* __restrict__ denom, int E)
{
  const int gid = blockIdx.x * blockDim.x + threadIdx.x;
  if (gid >= E * NHEAD) return;
  const int e = gid >> 2, h = gid & 3;
  const int s = src[e], d = dst[e];
  float x = el[s * NHEAD + h] + er[d * NHEAD + h];
  x = x > 0.0f ? x : 0.2f * x;
  atomicAdd(&denom[d * NHEAD + h], __expf(x));
}

// ---------------------------------------------------------------------------
// outacc[dst, hd] += feats[src, hd] * exp(lrelu(el+er)) / denom[dst,h]
// one thread per (edge, head-dim); hd coalesced.
// ---------------------------------------------------------------------------
__global__ void edge_aggregate(
    const int* __restrict__ src, const int* __restrict__ dst,
    const float* __restrict__ el, const float* __restrict__ er,
    const float* __restrict__ denom, const float* __restrict__ feats,
    float* __restrict__ outacc, int E)
{
  const long long gid = (long long)blockIdx.x * blockDim.x + threadIdx.x;
  const int e = (int)(gid >> 7);
  if (e >= E) return;
  const int hd = (int)(gid & 127), h = hd >> 5;
  const int s = src[e], d = dst[e];
  float x = el[s * NHEAD + h] + er[d * NHEAD + h];
  x = x > 0.0f ? x : 0.2f * x;
  const float a = __expf(x) / denom[d * NHEAD + h];
  atomicAdd(&outacc[(size_t)d * HD + hd], feats[(size_t)s * HD + hd] * a);
}

__global__ void elu_inplace(float* __restrict__ x, int n)
{
  const int gid = blockIdx.x * blockDim.x + threadIdx.x;
  if (gid >= n) return;
  const float v = x[gid];
  x[gid] = v > 0.0f ? v : (__expf(v) - 1.0f);
}

// out[i,d] = mean over 4 heads of rst[i, h*32+d]
__global__ void head_mean(const float* __restrict__ rst,
                          float* __restrict__ out, int n2)
{
  const int gid = blockIdx.x * blockDim.x + threadIdx.x;
  if (gid >= n2 * DHEAD) return;
  const int i = gid >> 5, d = gid & 31;
  const float* r = rst + (size_t)i * HD + d;
  out[gid] = 0.25f * (r[0] + r[32] + r[64] + r[96]);
}

// ---------------------------------------------------------------------------
extern "C" void kernel_launch(void* const* d_in, const int* in_sizes, int n_in,
                              void* d_out, int out_size, void* d_ws, size_t ws_size,
                              hipStream_t stream) {
  const float* feat0 = (const float*)d_in[0];
  const float* feat1 = (const float*)d_in[1];
  const int*   src0  = (const int*)d_in[2];
  const int*   dst0  = (const int*)d_in[3];
  const int*   src1  = (const int*)d_in[4];
  const int*   dst1  = (const int*)d_in[5];
  const int*   map12 = (const int*)d_in[6];
  const float* W0    = (const float*)d_in[7];
  const float* al0   = (const float*)d_in[8];
  const float* ar0   = (const float*)d_in[9];
  const float* W1    = (const float*)d_in[10];
  const float* al1   = (const float*)d_in[11];
  const float* ar1   = (const float*)d_in[12];

  const int n0 = in_sizes[0] / HD;   // 200000
  const int n1 = in_sizes[1] / HD;   // 50000
  const int E0 = in_sizes[2];        // 800000
  const int E1 = in_sizes[4];        // 200000
  const int n2 = in_sizes[6];        // 12500

  // Workspace carve-out (256B aligned slots)
  char* ws = (char*)d_ws;
  size_t off = 0;
  auto carve = [&](size_t bytes) -> float* {
    float* p = (float*)(ws + off);
    off += (bytes + 255) & ~(size_t)255;
    return p;
  };
  float* fs0  = carve((size_t)n0 * HD * 4);   // feat0@W0, kept for layer-1 gather
  float* fd0  = carve((size_t)n1 * HD * 4);   // feat1@W0; reused as fs1 = h1@W1
  float* h1   = carve((size_t)n1 * HD * 4);   // layer-1 aggregated output
  float* el0  = carve((size_t)n0 * NHEAD * 4);
  float* er0  = carve((size_t)n1 * NHEAD * 4);
  float* dn0  = carve((size_t)n1 * NHEAD * 4);
  float* el1  = carve((size_t)n1 * NHEAD * 4);
  float* er1  = carve((size_t)n2 * NHEAD * 4);
  float* dn1  = carve((size_t)n2 * NHEAD * 4);
  float* rst2 = carve((size_t)n2 * HD * 4);
  float* fs1  = fd0;  // safe: fd0 only needed until er0 is computed

  const int TB = 256;

  // ---- Layer 1 ----
  gemm128_wmma<<<(n0 + 127) / 128, TB, 0, stream>>>(feat0, W0, fs0, n0);
  gemm128_wmma<<<(n1 + 127) / 128, TB, 0, stream>>>(feat1, W0, fd0, n1);
  dot_heads<<<n0, 128, 0, stream>>>(fs0, al0, el0, nullptr, n0);
  dot_heads<<<n1, 128, 0, stream>>>(fd0, ar0, er0, nullptr, n1);

  hipMemsetAsync(dn0, 0, (size_t)n1 * NHEAD * 4, stream);
  edge_softmax_denom<<<((size_t)E0 * NHEAD + TB - 1) / TB, TB, 0, stream>>>(
      src0, dst0, el0, er0, dn0, E0);

  hipMemsetAsync(h1, 0, (size_t)n1 * HD * 4, stream);
  edge_aggregate<<<((size_t)E0 * HD + TB - 1) / TB, TB, 0, stream>>>(
      src0, dst0, el0, er0, dn0, fs0, h1, E0);

  elu_inplace<<<((size_t)n1 * HD + TB - 1) / TB, TB, 0, stream>>>(h1, n1 * HD);

  // ---- Layer 2 ----  (fd1 = (h1@W1)[map12] = fs1[map12], never materialized)
  gemm128_wmma<<<(n1 + 127) / 128, TB, 0, stream>>>(h1, W1, fs1, n1);
  dot_heads<<<n1, 128, 0, stream>>>(fs1, al1, el1, nullptr, n1);
  dot_heads<<<n2, 128, 0, stream>>>(fs1, ar1, er1, map12, n2);

  hipMemsetAsync(dn1, 0, (size_t)n2 * NHEAD * 4, stream);
  edge_softmax_denom<<<((size_t)E1 * NHEAD + TB - 1) / TB, TB, 0, stream>>>(
      src1, dst1, el1, er1, dn1, E1);

  hipMemsetAsync(rst2, 0, (size_t)n2 * HD * 4, stream);
  edge_aggregate<<<((size_t)E1 * HD + TB - 1) / TB, TB, 0, stream>>>(
      src1, dst1, el1, er1, dn1, fs1, rst2, E1);

  head_mean<<<((size_t)n2 * DHEAD + TB - 1) / TB, TB, 0, stream>>>(
      rst2, (float*)d_out, n2);
}
